// Net_89627377533404
// MI455X (gfx1250) — matrix operand order repaired and checked by reading, not compiled, not run on hardware
//
#include <hip/hip_runtime.h>
#include <hip/hip_fp16.h>

#define NPTS   1024
#define KNN    20
#define KS     7
#define FN     60
#define NC     40
#define BSZ    32
#define NTOT   (BSZ * NPTS)     // 32768
#define EPSBN  1e-5f

typedef __attribute__((ext_vector_type(16))) _Float16 v16h;
typedef __attribute__((ext_vector_type(8)))  float    v8f;

// ---------- monotone float<->uint map (for atomic max over signed floats) ----------
__device__ __forceinline__ unsigned fmapu(float x) {
  unsigned u = __float_as_uint(x);
  return (u & 0x80000000u) ? ~u : (u | 0x80000000u);
}
__device__ __forceinline__ float funmap(unsigned m) {
  return (m & 0x80000000u) ? __uint_as_float(m & 0x7fffffffu) : __uint_as_float(~m);
}

// =====================================================================
// 1) knn_graph: one wave per query row; distance row in LDS; 20 min passes
// =====================================================================
__global__ void knn_kernel(const float* __restrict__ pos, int* __restrict__ nbr) {
  __shared__ float sp[NPTS * 3];      // 12 KB
  __shared__ float srow[8][NPTS];     // 32 KB
  const int b = blockIdx.x >> 7;
  const int p0 = (blockIdx.x & 127) * 8;
  const float* pb = pos + (size_t)b * NPTS * 3;
  for (int i = threadIdx.x; i < NPTS * 3; i += 256) sp[i] = pb[i];
  __syncthreads();
  const int wave = threadIdx.x >> 5, lane = threadIdx.x & 31;
  const int p = p0 + wave;
  const float px = sp[p * 3], py = sp[p * 3 + 1], pz = sp[p * 3 + 2];
  const float sqp = px * px + py * py + pz * pz;
  for (int j = 0; j < 32; ++j) {
    int q = j * 32 + lane;
    float qx = sp[q * 3], qy = sp[q * 3 + 1], qz = sp[q * 3 + 2];
    float d = sqp + (qx * qx + qy * qy + qz * qz)
            - 2.f * (px * qx + py * qy + pz * qz);
    if (q == p) d = 1e10f;
    srow[wave][q] = d;
  }
  __syncthreads();
  for (int kk = 0; kk < KNN; ++kk) {
    float best = 3.0e38f; int bidx = 0;
    for (int j = 0; j < 32; ++j) {
      int q = j * 32 + lane;
      float d = srow[wave][q];
      if (d < best) { best = d; bidx = q; }
    }
    #pragma unroll
    for (int off = 16; off >= 1; off >>= 1) {
      float ov = __shfl_xor(best, off, 32);
      int   oi = __shfl_xor(bidx, off, 32);
      if (ov < best) { best = ov; bidx = oi; }
    }
    bidx = __shfl(bidx, 0, 32);
    if (lane == 0) {
      nbr[((size_t)b * NPTS + p) * KNN + kk] = b * NPTS + bidx;
      srow[wave][bidx] = 3.0e38f;
    }
    __syncthreads();
  }
}

// =====================================================================
// 2) DS spline layer: feat(n, c, d) + channel sums for BN
// =====================================================================
__global__ void ds_kernel(const float* __restrict__ pos, const int* __restrict__ nbr,
                          const float* __restrict__ Wsp,
                          float* __restrict__ feat, float* __restrict__ dsum,
                          float* __restrict__ dsq) {
  __shared__ float rel[KNN][3];
  __shared__ float dist[KNN];
  __shared__ float basis[KNN][KS];
  __shared__ float dmax;
  const int n = blockIdx.x;
  const int t = threadIdx.x;            // 192 threads
  if (t < KNN) {
    int g = nbr[(size_t)n * KNN + t];
    float rx = pos[(size_t)g * 3 + 0] - pos[(size_t)n * 3 + 0];
    float ry = pos[(size_t)g * 3 + 1] - pos[(size_t)n * 3 + 1];
    float rz = pos[(size_t)g * 3 + 2] - pos[(size_t)n * 3 + 2];
    rel[t][0] = rx; rel[t][1] = ry; rel[t][2] = rz;
    dist[t] = sqrtf(rx * rx + ry * ry + rz * rz + 1e-12f);
  }
  __syncthreads();
  if (t == 0) {
    float m = 0.f;
    for (int k = 0; k < KNN; ++k) m = fmaxf(m, dist[k]);
    dmax = m;
  }
  __syncthreads();
  if (t < KNN * KS) {
    int k = t / KS, m = t % KS;
    float u  = dist[k] / (dmax + 1e-9f);
    float tt = u * (float)(KS - 1);
    basis[k][m] = fmaxf(0.f, 1.f - fabsf(tt - (float)m));
  }
  __syncthreads();
  if (t < FN * 3) {
    int c = t / 3, d = t % 3;
    float acc = 0.f;
    for (int k = 0; k < KNN; ++k) {
      float w = 0.f;
      #pragma unroll
      for (int m = 0; m < KS; ++m) w += basis[k][m] * Wsp[m * FN + c];
      acc += w * rel[k][d];
    }
    acc *= (1.f / (float)KNN);
    feat[(size_t)n * (FN * 3) + t] = acc;
    atomicAdd(&dsum[c], acc);
    atomicAdd(&dsq[c], acc * acc);
  }
}

// =====================================================================
// 3) DS BN apply + pack to f16 (row-major, K padded 180 -> 192)
// =====================================================================
__global__ void ds_apply_kernel(const float* __restrict__ feat,
                                const float* __restrict__ dsum, const float* __restrict__ dsq,
                                const float* __restrict__ g, const float* __restrict__ bb,
                                _Float16* __restrict__ feath) {
  const int idx = blockIdx.x * 256 + threadIdx.x;   // N*192
  const int n = idx / 192, j = idx % 192;
  _Float16 v = (_Float16)0.f;
  if (j < FN * 3) {
    int c = j / 3;
    float cnt  = (float)NTOT * 3.f;
    float mean = dsum[c] / cnt;
    float var  = dsq[c] / cnt - mean * mean;
    float sc   = g[c] * rsqrtf(var + EPSBN);
    float x    = feat[(size_t)n * (FN * 3) + j];
    v = (_Float16)((x - mean) * sc + bb[c]);
  }
  feath[(size_t)n * 192 + j] = v;
}

// =====================================================================
// 4) Weight conversion: pack f16 weights in WMMA B-fragment order.
//    Fragment layout (wave32, f16 B 32x16): col = lane&15, k = (h&7) + 16*(h>>3) + 8*(lane>>4).
//    Packed index: (((tile*nChunk + kk)*32 + lane)*16 + h)
// =====================================================================
__global__ void convw_kernel(const float* __restrict__ ddW, const float* __restrict__ W1,
                             _Float16* __restrict__ ddWp, _Float16* __restrict__ W1p) {
  const int idx = blockIdx.x * 256 + threadIdx.x;
  if (idx < 192 * 128) {                    // ddW: 8 col-tiles x 6 k-chunks
    int h = idx & 15, cl = idx >> 4;
    int lane = cl & 31, tc = cl >> 5;
    int kk = tc % 6, t = tc / 6;
    int k   = kk * 32 + ((lane >> 4) << 3) + (h & 7) + ((h >> 3) << 4);
    int col = t * 16 + (lane & 15);
    ddWp[idx] = (k < 180) ? (_Float16)ddW[(size_t)k * 128 + col] : (_Float16)0.f;
  } else {                                  // W1: 64 col-tiles x 4 k-chunks
    int i2 = idx - 192 * 128;
    int h = i2 & 15, cl = i2 >> 4;
    int lane = cl & 31, tc = cl >> 5;
    int kk = tc & 3, t = tc >> 2;
    int k   = kk * 32 + ((lane >> 4) << 3) + (h & 7) + ((h >> 3) << 4);
    int col = t * 16 + (lane & 15);
    W1p[i2] = (_Float16)W1[(size_t)k * 1024 + col];
  }
}

// ---------- WMMA A-fragment loader from an LDS row (ISA 7.12.2, wave32) ----------
__device__ __forceinline__ v16h load_a_frag(const _Float16* rowptr, int lane, int kbase) {
  const int koff = ((lane >> 4) << 3) + kbase;
  v16h a;
  #pragma unroll
  for (int h = 0; h < 16; ++h) {
    int k = (h & 7) + ((h >> 3) << 4) + koff;   // two contiguous 8-half groups
    a[h] = rowptr[k];
  }
  return a;
}

// =====================================================================
// 5) DD edge-MLP GEMM via WMMA. A tile gathered with async global->LDS.
//    LDS row stride padded to 256 halves -> all addressing is shifts.
// =====================================================================
__global__ void dd_gemm_kernel(const _Float16* __restrict__ feath, const int* __restrict__ nbr,
                               const _Float16* __restrict__ ddWp, const float* __restrict__ ddb,
                               unsigned* __restrict__ hmax, unsigned* __restrict__ hmin,
                               float* __restrict__ ddsum, float* __restrict__ ddsq) {
  __shared__ _Float16 As[16][256];                 // 8 KB, rows padded to 512 B
  const int e0 = blockIdx.x * 16;
  const int n0 = e0 / KNN;                         // uniform -> SALU
  const int r0 = e0 - n0 * KNN;                    // e0 % 20
  // 16 rows x 24 16-byte chunks, gathered by the async DMA path (ASYNCcnt)
  #pragma unroll
  for (int it = 0; it < 2; ++it) {
    int ch = threadIdx.x + it * 256;               // 0..511
    int r = ch >> 5, j = ch & 31;                  // shifts, no division
    if (j < 24) {
      int src = nbr[e0 + r];                       // nbr flat (N*K): edge id == index
      unsigned lds  = (unsigned)(size_t)&As[r][j * 8];
      unsigned goff = (unsigned)src * 384u + (unsigned)j * 16u;
      asm volatile("global_load_async_to_lds_b128 %0, %1, %2"
                   :: "v"(lds), "v"(goff), "s"(feath) : "memory");
    }
  }
  asm volatile("s_wait_asynccnt 0x0" ::: "memory");
  __syncthreads();

  const int wave = threadIdx.x >> 5, lane = threadIdx.x & 31;
  const int col  = wave * 16 + (lane & 15);
  const int arow = lane & 15;
  const v16h* bp = (const v16h*)ddWp;
  v8f acc = {};
  #pragma unroll
  for (int kk = 0; kk < 6; ++kk) {
    v16h a = load_a_frag(&As[arow][0], lane, kk * 32);
    v16h b = bp[(wave * 6 + kk) * 32 + lane];      // pre-packed fragment: 2x b128
    acc = __builtin_amdgcn_wmma_f32_16x16x32_f16(false, a, false, b, (short)0, acc,
                                                 false, false);
  }
  const float bias = ddb[col];
  const int Mbase = (lane >> 4) << 3;
  float s = 0.f, sq = 0.f;
  #pragma unroll
  for (int r = 0; r < 8; ++r) {
    int M = r + Mbase;
    int n = n0 + ((r0 + M) >= KNN ? 1 : 0);        // replaces /20 magic division
    float v = fmaxf(acc[r] + bias, 0.f);           // relu -> bits monotone (>=0)
    s += v; sq += v * v;
    unsigned u = __float_as_uint(v);
    atomicMax(&hmax[(size_t)n * 128 + col], u);
    atomicMin(&hmin[(size_t)n * 128 + col], u);
  }
  s  += __shfl_xor(s, 16, 32);
  sq += __shfl_xor(sq, 16, 32);
  if (lane < 16) { atomicAdd(&ddsum[col], s); atomicAdd(&ddsq[col], sq); }
}

// =====================================================================
// 6) DD BN finalize + max-over-K; emit f2 directly in WMMA A-fragment order.
//    A-frag layout: row = lane&15, k = (h&7) + 16*(h>>3) + 8*(lane>>4).
// =====================================================================
__global__ void dd_fin_kernel(const unsigned* __restrict__ hmax, const unsigned* __restrict__ hmin,
                              const float* __restrict__ ddsum, const float* __restrict__ ddsq,
                              const float* __restrict__ g, const float* __restrict__ bb,
                              _Float16* __restrict__ f2p) {
  const int idx = blockIdx.x * 256 + threadIdx.x;   // N*128
  const int n = idx >> 7, c = idx & 127;
  const float cnt  = (float)NTOT * (float)KNN;
  const float mean = ddsum[c] / cnt;
  const float var  = ddsq[c] / cnt - mean * mean;
  const float sc   = g[c] * rsqrtf(var + EPSBN);
  const float sh   = bb[c] - mean * sc;
  const float vmax = __uint_as_float(hmax[idx]);
  const float vmin = __uint_as_float(hmin[idx]);
  const float val  = (sc >= 0.f) ? (sc * vmax + sh) : (sc * vmin + sh);
  // scatter into packed A-fragment layout (row-tile, k-chunk, lane, half)
  const int rt = n >> 4, rrow = n & 15;
  const int kk = c >> 5, kl = c & 31;
  const int lane = rrow + (((kl >> 3) & 1) << 4);
  const int h    = (kl & 7) + ((kl >> 4) << 3);
  f2p[(((size_t)rt * 4 + kk) * 32 + lane) * 16 + h] = (_Float16)val;
}

// =====================================================================
// 7) Head y1 GEMM via WMMA: both operands pre-packed -> pure b128 + wmma
// =====================================================================
__global__ void y1_gemm_kernel(const _Float16* __restrict__ f2p, const _Float16* __restrict__ W1p,
                               const float* __restrict__ b1, unsigned* __restrict__ y1max) {
  const int wave = threadIdx.x >> 5, lane = threadIdx.x & 31;
  const int rt = blockIdx.x >> 3;                       // row tile (16 points)
  const int tg = (blockIdx.x & 7) * 8 + wave;           // col tile (of 64)
  const v16h* ap = (const v16h*)f2p;
  const v16h* bp = (const v16h*)W1p;
  v8f acc = {};
  #pragma unroll
  for (int kk = 0; kk < 4; ++kk) {
    v16h a = ap[((size_t)rt * 4 + kk) * 32 + lane];
    v16h b = bp[((size_t)tg * 4 + kk) * 32 + lane];
    acc = __builtin_amdgcn_wmma_f32_16x16x32_f16(false, a, false, b, (short)0, acc,
                                                 false, false);
  }
  const int col  = tg * 16 + (lane & 15);
  const float bias = b1[col];
  const int Mbase = (lane >> 4) << 3;
  #pragma unroll
  for (int r = 0; r < 8; ++r) {
    int smp = (rt * 16 + r + Mbase) >> 10;
    atomicMax(&y1max[(size_t)smp * 1024 + col], fmapu(acc[r] + bias));
  }
}

// =====================================================================
// 8) y1: unmap, relu, BN over batch (count=32) -> y1bn
// =====================================================================
__global__ void y1_bn_kernel(const unsigned* __restrict__ y1max,
                             const float* __restrict__ g, const float* __restrict__ bb,
                             float* __restrict__ y1bn) {
  const int ch = blockIdx.x * 256 + threadIdx.x;    // 1024 channels
  float vals[BSZ];
  float s = 0.f, sq = 0.f;
  #pragma unroll
  for (int smp = 0; smp < BSZ; ++smp) {
    float v = fmaxf(funmap(y1max[smp * 1024 + ch]), 0.f);
    vals[smp] = v; s += v; sq += v * v;
  }
  float mean = s / (float)BSZ;
  float var  = sq / (float)BSZ - mean * mean;
  float sc   = g[ch] * rsqrtf(var + EPSBN);
  #pragma unroll
  for (int smp = 0; smp < BSZ; ++smp)
    y1bn[smp * 1024 + ch] = (vals[smp] - mean) * sc + bb[ch];
}

// =====================================================================
// 9/11) small GEMM + relu + channel stats (tail layers, VALU)
// =====================================================================
__global__ void gemm_relu_stats_kernel(const float* __restrict__ X, const float* __restrict__ W,
                                       const float* __restrict__ bias, float* __restrict__ raw,
                                       float* __restrict__ csum, float* __restrict__ csq,
                                       int Kdim, int C, int total) {
  const int idx = blockIdx.x * 256 + threadIdx.x;
  if (idx >= total) return;
  const int s = idx / C, c = idx - s * C;
  float acc = bias[c];
  for (int k = 0; k < Kdim; ++k) acc += X[(size_t)s * Kdim + k] * W[(size_t)k * C + c];
  float z = fmaxf(acc, 0.f);
  raw[idx] = z;
  atomicAdd(&csum[c], z);
  atomicAdd(&csq[c], z * z);
}

// 10/12) BN apply over batch axis (count = BSZ)
__global__ void bn_apply_kernel(const float* __restrict__ raw, const float* __restrict__ csum,
                                const float* __restrict__ csq, const float* __restrict__ g,
                                const float* __restrict__ bb, float* __restrict__ out,
                                int C, int total) {
  const int idx = blockIdx.x * 256 + threadIdx.x;
  if (idx >= total) return;
  const int c = idx % C;
  float mean = csum[c] / (float)BSZ;
  float var  = csq[c] / (float)BSZ - mean * mean;
  float sc   = g[c] * rsqrtf(var + EPSBN);
  out[idx] = (raw[idx] - mean) * sc + bb[c];
}

// =====================================================================
// 13) y4 + log_softmax
// =====================================================================
__global__ void head_final_kernel(const float* __restrict__ y3bn, const float* __restrict__ W4,
                                  const float* __restrict__ b4, float* __restrict__ out) {
  __shared__ float v[NC];
  const int row = blockIdx.x, t = threadIdx.x;
  if (t < NC) {
    float acc = b4[t];
    for (int k = 0; k < 265; ++k) acc += y3bn[(size_t)row * 265 + k] * W4[(size_t)k * NC + t];
    v[t] = acc;
  }
  __syncthreads();
  if (t < NC) {
    float m = -3e38f;
    for (int c = 0; c < NC; ++c) m = fmaxf(m, v[c]);
    float se = 0.f;
    for (int c = 0; c < NC; ++c) se += expf(v[c] - m);
    out[(size_t)row * NC + t] = v[t] - m - logf(se);
  }
}

// =====================================================================
// Host orchestration
// =====================================================================
extern "C" void kernel_launch(void* const* d_in, const int* in_sizes, int n_in,
                              void* d_out, int out_size, void* d_ws, size_t ws_size,
                              hipStream_t stream) {
  const float* pos    = (const float*)d_in[0];
  const float* Wsp    = (const float*)d_in[2];
  const float* ds_g   = (const float*)d_in[3];
  const float* ds_b   = (const float*)d_in[4];
  const float* ddW    = (const float*)d_in[5];
  const float* ddb    = (const float*)d_in[6];
  const float* dd_g   = (const float*)d_in[7];
  const float* dd_bb  = (const float*)d_in[8];
  const float* W1     = (const float*)d_in[9];
  const float* b1     = (const float*)d_in[10];
  const float* g1     = (const float*)d_in[11];
  const float* bb1    = (const float*)d_in[12];
  const float* W2     = (const float*)d_in[13];
  const float* b2     = (const float*)d_in[14];
  const float* g2     = (const float*)d_in[15];
  const float* bb2    = (const float*)d_in[16];
  const float* W3     = (const float*)d_in[17];
  const float* b3     = (const float*)d_in[18];
  const float* g3     = (const float*)d_in[19];
  const float* bb3    = (const float*)d_in[20];
  const float* W4     = (const float*)d_in[21];
  const float* b4     = (const float*)d_in[22];
  float* out = (float*)d_out;

  char* ws = (char*)d_ws;
  size_t off = 0;
  auto alloc = [&](size_t bytes) -> void* {
    void* p = ws + off;
    off = (off + bytes + 255) & ~(size_t)255;
    return p;
  };
  float* stats  = (float*)alloc(1984 * sizeof(float));
  float* dsum   = stats;         float* dsq  = stats + 64;
  float* ddsum  = stats + 128;   float* ddsq = stats + 256;
  float* csum2  = stats + 384;   float* csq2 = stats + 896;
  float* csum3  = stats + 1408;  float* csq3 = stats + 1696;

  int*      nbr   = (int*)     alloc((size_t)NTOT * KNN * 4);
  float*    feat  = (float*)   alloc((size_t)NTOT * FN * 3 * 4);
  _Float16* feath = (_Float16*)alloc((size_t)NTOT * 192 * 2);
  _Float16* ddWp  = (_Float16*)alloc((size_t)192 * 128 * 2);
  _Float16* W1p   = (_Float16*)alloc((size_t)128 * 1024 * 2);
  unsigned* hmax  = (unsigned*)alloc((size_t)NTOT * 128 * 4);
  unsigned* hmin  = (unsigned*)alloc((size_t)NTOT * 128 * 4);
  _Float16* f2p   = (_Float16*)alloc((size_t)NTOT * 128 * 2);
  unsigned* y1max = (unsigned*)alloc((size_t)BSZ * 1024 * 4);
  float*    y1bn  = (float*)   alloc((size_t)BSZ * 1024 * 4);
  float*    y2raw = (float*)   alloc((size_t)BSZ * 512 * 4);
  float*    y2bn  = (float*)   alloc((size_t)BSZ * 512 * 4);
  float*    y3raw = (float*)   alloc((size_t)BSZ * 265 * 4);
  float*    y3bn  = (float*)   alloc((size_t)BSZ * 265 * 4);

  hipMemsetAsync(stats, 0,    1984 * sizeof(float),   stream);
  hipMemsetAsync(hmax,  0,    (size_t)NTOT * 128 * 4, stream);
  hipMemsetAsync(hmin,  0x7F, (size_t)NTOT * 128 * 4, stream);
  hipMemsetAsync(y1max, 0,    (size_t)BSZ * 1024 * 4, stream);

  knn_kernel<<<4096, 256, 0, stream>>>(pos, nbr);
  ds_kernel<<<NTOT, 192, 0, stream>>>(pos, nbr, Wsp, feat, dsum, dsq);
  ds_apply_kernel<<<(NTOT * 192) / 256, 256, 0, stream>>>(feat, dsum, dsq, ds_g, ds_b, feath);
  convw_kernel<<<(192 * 128 + 128 * 1024) / 256, 256, 0, stream>>>(ddW, W1, ddWp, W1p);
  dd_gemm_kernel<<<(NTOT * KNN) / 16, 256, 0, stream>>>(feath, nbr, ddWp, ddb,
                                                        hmax, hmin, ddsum, ddsq);
  dd_fin_kernel<<<(NTOT * 128) / 256, 256, 0, stream>>>(hmax, hmin, ddsum, ddsq,
                                                        dd_g, dd_bb, f2p);
  y1_gemm_kernel<<<(NTOT / 16) * 8, 256, 0, stream>>>(f2p, W1p, b1, y1max);
  y1_bn_kernel<<<4, 256, 0, stream>>>(y1max, g1, bb1, y1bn);
  gemm_relu_stats_kernel<<<(BSZ * 512 + 255) / 256, 256, 0, stream>>>(
      y1bn, W2, b2, y2raw, csum2, csq2, 1024, 512, BSZ * 512);
  bn_apply_kernel<<<(BSZ * 512 + 255) / 256, 256, 0, stream>>>(
      y2raw, csum2, csq2, g2, bb2, y2bn, 512, BSZ * 512);
  gemm_relu_stats_kernel<<<(BSZ * 265 + 255) / 256, 256, 0, stream>>>(
      y2bn, W3, b3, y3raw, csum3, csq3, 512, 265, BSZ * 265);
  bn_apply_kernel<<<(BSZ * 265 + 255) / 256, 256, 0, stream>>>(
      y3raw, csum3, csq3, g3, bb3, y3bn, 265, BSZ * 265);
  head_final_kernel<<<BSZ, 64, 0, stream>>>(y3bn, W4, b4, out);

  (void)in_sizes; (void)n_in; (void)out_size; (void)ws_size;
}
